// Main_model_49323404427799
// MI455X (gfx1250) — compile-verified, matrix operand order checked by reading, not run on hardware
//
#include <hip/hip_runtime.h>
#include <math.h>

typedef __attribute__((ext_vector_type(16))) __bf16 v16bf;
typedef __attribute__((ext_vector_type(8)))  __bf16 v8bf;
typedef __attribute__((ext_vector_type(8)))  float  v8f;

#define HID 128

__device__ inline float nan0(float v) {
  if (!(v == v)) return 0.f;
  if (v >  3.402823466e38f) return  3.402823466e38f;
  if (v < -3.402823466e38f) return -3.402823466e38f;
  return v;
}

// ---------------- f32 [128 x 128] (K-major) -> bf16 transposed [N][K] ----------------
__global__ void k_cvt_bf16_t128(const float* __restrict__ src, __bf16* __restrict__ dst) {
  int i = blockIdx.x * 256 + threadIdx.x;  // 16384 total
  int k = i >> 7, n = i & 127;
  dst[n * HID + k] = (__bf16)src[k * HID + n];
}

// ---------------- zero buffer ----------------
__global__ void k_zero(float* __restrict__ p, int n) {
  int i = blockIdx.x * 256 + threadIdx.x;
  if (i < n) p[i] = 0.f;
}

// ---------------- gather embeddings (atom part, fp part) as bf16 ----------------
__global__ void k_gather(const int* __restrict__ atom_ids, const int* __restrict__ fp_ids,
                         const float* __restrict__ emb_atom, const float* __restrict__ emb_fp,
                         __bf16* __restrict__ xa, __bf16* __restrict__ xf) {
  int n = blockIdx.x;
  int c = threadIdx.x;  // 128
  xa[(size_t)n * HID + c] = (__bf16)emb_atom[(size_t)atom_ids[n] * HID + c];
  xf[(size_t)n * HID + c] = (__bf16)emb_fp[(size_t)fp_ids[n] * HID + c];
}

__device__ inline v16bf load_wmma_operand(const __bf16* p) {
  // 16 elements = two contiguous 8-element (16B) runs, 16B apart
  v8bf lo = *(const v8bf*)p;
  v8bf hi = *(const v8bf*)(p + 16);
  v16bf r;
#pragma unroll
  for (int j = 0; j < 8; ++j) { r[j] = lo[j]; r[8 + j] = hi[j]; }
  return r;
}

// ---------------- WMMA GEMM: out[M,128] = A[M,128](bf16) @ W + bias (+resid) ----------------
// Bt is the weight matrix TRANSPOSED: Bt[n*128 + k] = W[k][n]  (bf16)
// Compile-time config -> straight-line epilogue.
template <bool RELU, bool NAN2, bool HAS_BIAS, bool HAS_RESID, bool OUTF, bool OUTBF>
__global__ __launch_bounds__(256) void k_gemm128(
    const __bf16* __restrict__ A, const __bf16* __restrict__ Bt,
    const float* __restrict__ bias, const float* __restrict__ resid,
    float* __restrict__ outf, __bf16* __restrict__ outbf, int M) {
  __shared__ __align__(16) __bf16 Ash[16 * HID];
  __shared__ __align__(16) __bf16 Bsh[HID * HID];
  const int tid = threadIdx.x;
  const int tile = blockIdx.x;
  // stage A tile (16x128 bf16 = 256 uint4) and full Bt (128x128 bf16 = 2048 uint4)
  {
    const uint4* ag = (const uint4*)(A + (size_t)tile * 16 * HID);
    uint4* as = (uint4*)Ash;
    as[tid] = ag[tid];
    const uint4* bg = (const uint4*)Bt;
    uint4* bs = (uint4*)Bsh;
#pragma unroll
    for (int i = 0; i < 8; ++i) bs[tid + 256 * i] = bg[tid + 256 * i];
  }
  __syncthreads();
  const int wave = tid >> 5;
  const int lane = tid & 31;
  const int r16 = lane & 15;
  const int khalf = lane >> 4;
  const int ncol = wave * 16 + r16;
  v8f acc = {0.f, 0.f, 0.f, 0.f, 0.f, 0.f, 0.f, 0.f};
#pragma unroll
  for (int kk = 0; kk < HID; kk += 32) {
    // A: lane holds row (lane&15); K runs [kk+khalf*8 ..+7] and [kk+16+khalf*8 ..+7]
    v16bf a = load_wmma_operand(Ash + r16 * HID + kk + khalf * 8);
    // B: identical pattern on transposed weights, row = output column
    v16bf b = load_wmma_operand(Bsh + ncol * HID + kk + khalf * 8);
    acc = __builtin_amdgcn_wmma_f32_16x16x32_bf16(false, a, false, b, (short)0, acc,
                                                  false, false);
  }
  const int rowBase = tile * 16 + khalf * 8;
  const float bval = HAS_BIAS ? bias[ncol] : 0.f;
#pragma unroll
  for (int r = 0; r < 8; ++r) {
    const int row = rowBase + r;
    float v = acc[r] + bval;
    if (HAS_RESID) v += resid[(size_t)row * HID + ncol];
    if (RELU) v = fmaxf(v, 0.f);
    if (NAN2) v = nan0(v);
    if (OUTF)  outf[(size_t)row * HID + ncol] = v;
    if (OUTBF) outbf[(size_t)row * HID + ncol] = (__bf16)v;
  }
}

// ---------------- GINE message passing: agg[dst] += relu(h[src] + emb_bond[eid]) ----------------
__global__ void k_edges(const int* __restrict__ ei, const int* __restrict__ eattr,
                        const float* __restrict__ h, const float* __restrict__ emb_bond,
                        float* __restrict__ agg, int E) {
  long long g = (long long)blockIdx.x * 256 + threadIdx.x;
  int e = (int)(g >> 5);
  if (e >= E) return;
  int j = ((int)g & 31) * 4;
  int s = ei[e], d = ei[E + e];
  int b = eattr[e]; b = b < 0 ? 0 : (b > 7 ? 7 : b);
  const float4 hv = *(const float4*)(h + (size_t)s * HID + j);
  const float4 bv = *(const float4*)(emb_bond + (size_t)b * HID + j);
  float* o = agg + (size_t)d * HID + j;
  atomicAdd(o + 0, fmaxf(hv.x + bv.x, 0.f));
  atomicAdd(o + 1, fmaxf(hv.y + bv.y, 0.f));
  atomicAdd(o + 2, fmaxf(hv.z + bv.z, 0.f));
  atomicAdd(o + 3, fmaxf(hv.w + bv.w, 0.f));
}

__global__ void k_self_loops(const float* __restrict__ h, const float* __restrict__ emb_bond,
                             float* __restrict__ agg, int N) {
  long long g = (long long)blockIdx.x * 256 + threadIdx.x;
  int n = (int)(g >> 5);
  if (n >= N) return;
  int j = ((int)g & 31) * 4;
  const float4 hv = *(const float4*)(h + (size_t)n * HID + j);
  const float4 bv = *(const float4*)(emb_bond + j);  // bond id 0
  float* o = agg + (size_t)n * HID + j;
  o[0] += fmaxf(hv.x + bv.x, 0.f);
  o[1] += fmaxf(hv.y + bv.y, 0.f);
  o[2] += fmaxf(hv.z + bv.z, 0.f);
  o[3] += fmaxf(hv.w + bv.w, 0.f);
}

// ---------------- zbf = bf16(h + agg) ----------------
__global__ void k_addbf(const float* __restrict__ h, const float* __restrict__ agg,
                        __bf16* __restrict__ zbf, int n) {
  int i = blockIdx.x * 256 + threadIdx.x;
  if (i < n) zbf[i] = (__bf16)(h[i] + agg[i]);
}

// ---------------- global LayerNorm stats ----------------
__global__ void k_reduce_sums(const float* __restrict__ x, float* __restrict__ red, int n) {
  __shared__ float sh[256];
  __shared__ float sh2[256];
  float s = 0.f, s2 = 0.f;
  for (int i = blockIdx.x * 256 + threadIdx.x; i < n; i += 256 * gridDim.x) {
    float v = x[i]; s += v; s2 += v * v;
  }
  sh[threadIdx.x] = s; sh2[threadIdx.x] = s2;
  __syncthreads();
  for (int st = 128; st > 0; st >>= 1) {
    if (threadIdx.x < st) { sh[threadIdx.x] += sh[threadIdx.x + st]; sh2[threadIdx.x] += sh2[threadIdx.x + st]; }
    __syncthreads();
  }
  if (threadIdx.x == 0) { red[blockIdx.x] = sh[0]; red[gridDim.x + blockIdx.x] = sh2[0]; }
}

__global__ void k_final_stats(const float* __restrict__ red, float* __restrict__ stats,
                              int nblk, float inv_n) {
  __shared__ float sh[256];
  __shared__ float sh2[256];
  float s = 0.f, s2 = 0.f;
  for (int i = threadIdx.x; i < nblk; i += 256) { s += red[i]; s2 += red[nblk + i]; }
  sh[threadIdx.x] = s; sh2[threadIdx.x] = s2;
  __syncthreads();
  for (int st = 128; st > 0; st >>= 1) {
    if (threadIdx.x < st) { sh[threadIdx.x] += sh[threadIdx.x + st]; sh2[threadIdx.x] += sh2[threadIdx.x + st]; }
    __syncthreads();
  }
  if (threadIdx.x == 0) {
    float mu = sh[0] * inv_n;
    float var = fmaxf(sh2[0] * inv_n - mu * mu, 0.f);
    stats[0] = mu;
    stats[1] = rsqrtf(var + 1e-5f);
  }
}

// ---------------- LN + SiLU + nan_to_num, write f32 + bf16 ----------------
__global__ void k_ln_silu(const float* __restrict__ z, const float* __restrict__ stats,
                          const float* __restrict__ lnw, const float* __restrict__ lnb,
                          float* __restrict__ h, __bf16* __restrict__ hbf, int n) {
  int i = blockIdx.x * 256 + threadIdx.x;
  if (i >= n) return;
  int c = i & (HID - 1);
  float y = (z[i] - stats[0]) * stats[1] * lnw[c] + lnb[c];
  float s = y / (1.f + expf(-y));
  s = nan0(s);
  h[i] = s;
  hbf[i] = (__bf16)s;
}

// ---------------- column means of h (query vector source) ----------------
__global__ void k_hmean(const float* __restrict__ h, float* __restrict__ hmean, int N) {
  __shared__ float sh[256];
  int c = blockIdx.x;
  float s = 0.f;
  for (int n = threadIdx.x; n < N; n += 256) s += h[(size_t)n * HID + c];
  sh[threadIdx.x] = s;
  __syncthreads();
  for (int st = 128; st > 0; st >>= 1) {
    if (threadIdx.x < st) sh[threadIdx.x] += sh[threadIdx.x + st];
    __syncthreads();
  }
  if (threadIdx.x == 0) hmean[c] = sh[0] / (float)N;
}

// ---------------- qvec = hmean @ wq + bq ----------------
__global__ void k_qvec(const float* __restrict__ hmean, const float* __restrict__ wq,
                       const float* __restrict__ bq, float* __restrict__ qvec) {
  int j = threadIdx.x;  // 128
  float v = bq[j];
  for (int c = 0; c < HID; ++c) v += hmean[c] * wq[c * HID + j];
  qvec[j] = v;
}

// ---------------- kqv[head][c] = sum_d wk[c, 32h+d] * qvec[32h+d] ; kqb[head] ----------------
__global__ void k_kq(const float* __restrict__ wk, const float* __restrict__ bk,
                     const float* __restrict__ qvec, float* __restrict__ kqv,
                     float* __restrict__ kqb) {
  int t = threadIdx.x;  // 512
  int hd = t >> 7, c = t & 127;
  float v = 0.f;
  for (int d = 0; d < 32; ++d) v += wk[c * HID + hd * 32 + d] * qvec[hd * 32 + d];
  kqv[t] = v;
  if (t < 4) {
    float b = 0.f;
    for (int d = 0; d < 32; ++d) b += bk[t * 32 + d] * qvec[t * 32 + d];
    kqb[t] = b;
  }
}

// ---------------- qk[n,head] = (h[n].kqv[head] + kqb[head]) * scale ----------------
__global__ void k_qk(const float* __restrict__ h, const float* __restrict__ kqv,
                     const float* __restrict__ kqb, float* __restrict__ qk, int N) {
  int t = blockIdx.x * 256 + threadIdx.x;
  if (t >= N * 4) return;
  int n = t >> 2, hd = t & 3;
  const float* kv = kqv + hd * HID;
  const float* hp = h + (size_t)n * HID;
  float v = kqb[hd];
  for (int c = 0; c < HID; ++c) v += hp[c] * kv[c];
  qk[t] = v * 0.2209708691207961f;  // 1/(sqrt(32)*0.8)
}

// ---------------- per-subgraph masked softmax over nodes, weighted V sum ----------------
__global__ __launch_bounds__(128) void k_attn_pool(const unsigned char* __restrict__ mask,
                                                   const float* __restrict__ qk,
                                                   const float* __restrict__ V,
                                                   float* __restrict__ sub, int N) {
  const int s = blockIdx.x;
  const int t = threadIdx.x;  // 128
  const unsigned char* mk = mask + (size_t)s * N;
  __shared__ float red[128];
  __shared__ float mh[4], Zh[4], wU[4];
  float lmax[4] = {-3.0e38f, -3.0e38f, -3.0e38f, -3.0e38f};
  for (int n = t; n < N; n += 128) {
    if (mk[n]) {
      const float* q = qk + (size_t)n * 4;
      lmax[0] = fmaxf(lmax[0], q[0]); lmax[1] = fmaxf(lmax[1], q[1]);
      lmax[2] = fmaxf(lmax[2], q[2]); lmax[3] = fmaxf(lmax[3], q[3]);
    }
  }
  for (int hd = 0; hd < 4; ++hd) {
    red[t] = lmax[hd];
    __syncthreads();
    for (int st = 64; st > 0; st >>= 1) {
      if (t < st) red[t] = fmaxf(red[t], red[t + st]);
      __syncthreads();
    }
    if (t == 0) {
      float m = fmaxf(red[0], -1e9f);   // unmasked nodes score -1e9
      mh[hd] = m;
      wU[hd] = expf(-1e9f - m);         // 0 normally; 1 if nothing masked
    }
    __syncthreads();
  }
  float lsum[4] = {0.f, 0.f, 0.f, 0.f};
  for (int n = t; n < N; n += 128) {
    bool b = mk[n] != 0;
    const float* q = qk + (size_t)n * 4;
    for (int hd = 0; hd < 4; ++hd) lsum[hd] += b ? expf(q[hd] - mh[hd]) : wU[hd];
  }
  for (int hd = 0; hd < 4; ++hd) {
    red[t] = lsum[hd];
    __syncthreads();
    for (int st = 64; st > 0; st >>= 1) {
      if (t < st) red[t] += red[t + st];
      __syncthreads();
    }
    if (t == 0) Zh[hd] = red[0];
    __syncthreads();
  }
  const int hd = t >> 5;
  const float m = mh[hd], wu = wU[hd];
  float acc = 0.f;
  for (int n = 0; n < N; ++n) {
    unsigned char b = mk[n];
    if (b)               acc += expf(qk[(size_t)n * 4 + hd] - m) * V[(size_t)n * HID + t];
    else if (wu != 0.f)  acc += wu * V[(size_t)n * HID + t];
  }
  sub[(size_t)s * HID + t] = nan0(acc / Zh[hd]);
}

// ---------------- out projection + subgraph scores ----------------
__global__ __launch_bounds__(128) void k_out_proj(const float* __restrict__ sub,
                                                  const float* __restrict__ wo,
                                                  const float* __restrict__ bo,
                                                  const float* __restrict__ sgq,
                                                  float* __restrict__ sub2,
                                                  float* __restrict__ sg) {
  int s = blockIdx.x, j = threadIdx.x;  // 128
  __shared__ float srow[128];
  __shared__ float sh[128];
  srow[j] = sub[(size_t)s * HID + j];
  __syncthreads();
  float v = bo[j];
  for (int c = 0; c < HID; ++c) v += srow[c] * wo[c * HID + j];
  v = nan0(v);
  sub2[(size_t)s * HID + j] = v;
  sh[j] = v * sgq[j];
  __syncthreads();
  for (int st = 64; st > 0; st >>= 1) {
    if (j < st) sh[j] += sh[j + st];
    __syncthreads();
  }
  if (j == 0) sg[s] = nan0(sh[0] * 0.08838834764831845f);  // 1/sqrt(128)
}

// ---------------- top-4 + softmax + weighted sum -> mol_embed[128] ----------------
__global__ __launch_bounds__(128) void k_topk_out(const float* __restrict__ sub2,
                                                  const float* __restrict__ sg,
                                                  float* __restrict__ out) {
  __shared__ float vals[256];
  __shared__ int idx[4];
  __shared__ float w[4];
  int t = threadIdx.x;  // 128
  vals[t] = sg[t];
  vals[t + 128] = sg[t + 128];
  __syncthreads();
  if (t == 0) {
    float tv[4];
    for (int k = 0; k < 4; ++k) {
      float best = -3.4e38f; int bi = 0;
      for (int i = 0; i < 256; ++i) if (vals[i] > best) { best = vals[i]; bi = i; }
      tv[k] = best; idx[k] = bi; vals[bi] = -3.4e38f;
    }
    float mx = tv[0], ssum = 0.f;
    for (int k = 0; k < 4; ++k) { w[k] = expf(tv[k] - mx); ssum += w[k]; }
    for (int k = 0; k < 4; ++k) w[k] /= ssum;
  }
  __syncthreads();
  float v = 0.f;
  for (int k = 0; k < 4; ++k) v += w[k] * sub2[(size_t)idx[k] * HID + t];
  out[t] = v;
}

extern "C" void kernel_launch(void* const* d_in, const int* in_sizes, int n_in,
                              void* d_out, int out_size, void* d_ws, size_t ws_size,
                              hipStream_t stream) {
  const int* atom_ids = (const int*)d_in[0];
  const int* fp_ids   = (const int*)d_in[1];
  const int* edge_index = (const int*)d_in[2];
  const int* edge_attr  = (const int*)d_in[3];
  const unsigned char* sub_masks = (const unsigned char*)d_in[4];
  const float* emb_atom = (const float*)d_in[5];
  const float* emb_fp   = (const float*)d_in[6];
  const float* emb_bond = (const float*)d_in[7];
  const float* w_proj = (const float*)d_in[8];
  const float* b_proj = (const float*)d_in[9];
  const float* mlp_w1 = (const float*)d_in[10];
  const float* mlp_b1 = (const float*)d_in[11];
  const float* mlp_w2 = (const float*)d_in[12];
  const float* mlp_b2 = (const float*)d_in[13];
  const float* ln_w = (const float*)d_in[14];
  const float* ln_b = (const float*)d_in[15];
  const float* wq = (const float*)d_in[16];
  const float* bq = (const float*)d_in[17];
  const float* wk = (const float*)d_in[18];
  const float* bk = (const float*)d_in[19];
  const float* wv = (const float*)d_in[20];
  const float* bv = (const float*)d_in[21];
  const float* wo = (const float*)d_in[22];
  const float* bo = (const float*)d_in[23];
  const float* sgq = (const float*)d_in[24];

  const int N = in_sizes[0];          // 50000
  const int E = in_sizes[3];          // 800000
  const int NB = N * HID;             // 6,400,000
  const int NLAYERS = in_sizes[10] / (HID * HID);  // 3

  // ---- workspace carve ----
  char* base = (char*)d_ws;
  size_t off = 0;
  auto carve = [&](size_t bytes) -> void* {
    void* p = base + off;
    off = (off + bytes + 255) & ~(size_t)255;
    return p;
  };
  float*  h    = (float*)carve((size_t)NB * 4);
  float*  tmpf = (float*)carve((size_t)NB * 4);   // also aliases xa/xf bf16 regions
  __bf16* zbf  = (__bf16*)carve((size_t)NB * 2);
  __bf16* tbf  = (__bf16*)carve((size_t)NB * 2);
  __bf16* hbf  = (__bf16*)carve((size_t)NB * 2);
  float*  agg  = (float*)carve((size_t)NB * 4);
  float*  Vbuf = (float*)carve((size_t)NB * 4);
  __bf16* wbf  = (__bf16*)carve((size_t)(2 * HID * HID + 6 * HID * HID + HID * HID) * 2);
  float*  qk   = (float*)carve((size_t)N * 4 * 4);
  float*  red  = (float*)carve(2048 * 4);
  float*  stats = (float*)carve(8 * 4);
  float*  hmean = (float*)carve(HID * 4);
  float*  qvec  = (float*)carve(HID * 4);
  float*  kqv   = (float*)carve(4 * HID * 4);
  float*  kqb   = (float*)carve(4 * 4);
  float*  sub   = (float*)carve(256 * HID * 4);
  float*  sub2  = (float*)carve(256 * HID * 4);
  float*  sg    = (float*)carve(256 * 4);
  (void)ws_size; (void)n_in; (void)out_size;

  __bf16* xa = (__bf16*)tmpf;        // N*128 bf16
  __bf16* xf = xa + (size_t)NB;      // N*128 bf16
  // transposed bf16 weight blocks, each 128x128 stored as Bt[n][k]
  __bf16* wpbf = wbf;                         // 2 blocks (proj rows 0..127, 128..255)
  __bf16* w1bf = wbf + 2 * HID * HID;         // 3 blocks
  __bf16* w2bf = w1bf + 3 * HID * HID;        // 3 blocks
  __bf16* wvbf = w2bf + 3 * HID * HID;        // 1 block

  const int tiles = (N + 15) / 16;
  const int TBLK = (HID * HID) / 256;  // 64 blocks per 128x128 transpose

  // weight conversion + transpose into WMMA-friendly N-major layout
  k_cvt_bf16_t128<<<TBLK, 256, 0, stream>>>(w_proj, wpbf);
  k_cvt_bf16_t128<<<TBLK, 256, 0, stream>>>(w_proj + HID * HID, wpbf + HID * HID);
  for (int l = 0; l < NLAYERS; ++l) {
    k_cvt_bf16_t128<<<TBLK, 256, 0, stream>>>(mlp_w1 + l * HID * HID, w1bf + l * HID * HID);
    k_cvt_bf16_t128<<<TBLK, 256, 0, stream>>>(mlp_w2 + l * HID * HID, w2bf + l * HID * HID);
  }
  k_cvt_bf16_t128<<<TBLK, 256, 0, stream>>>(wv, wvbf);

  // input projection: h = nan_to_num([xa|xf] @ w_proj + b_proj), split K=256 into 2x K=128
  k_gather<<<N, 128, 0, stream>>>(atom_ids, fp_ids, emb_atom, emb_fp, xa, xf);
  // <RELU, NAN2, BIAS, RESID, OUTF, OUTBF>
  k_gemm128<false, false, true, false, true, false><<<tiles, 256, 0, stream>>>(
      xa, wpbf, b_proj, nullptr, h, nullptr, N);
  k_gemm128<false, true, false, true, true, true><<<tiles, 256, 0, stream>>>(
      xf, wpbf + HID * HID, nullptr, h, h, hbf, N);

  // GINE layers
  for (int l = 0; l < NLAYERS; ++l) {
    k_zero<<<(NB + 255) / 256, 256, 0, stream>>>(agg, NB);
    k_edges<<<(int)(((long long)E * 32 + 255) / 256), 256, 0, stream>>>(
        edge_index, edge_attr, h, emb_bond, agg, E);
    k_self_loops<<<(int)(((long long)N * 32 + 255) / 256), 256, 0, stream>>>(
        h, emb_bond, agg, N);
    k_addbf<<<(NB + 255) / 256, 256, 0, stream>>>(h, agg, zbf, NB);
    k_gemm128<true, false, true, false, false, true><<<tiles, 256, 0, stream>>>(
        zbf, w1bf + l * HID * HID, mlp_b1 + l * HID, nullptr, nullptr, tbf, N);
    k_gemm128<false, false, true, true, true, false><<<tiles, 256, 0, stream>>>(
        tbf, w2bf + l * HID * HID, mlp_b2 + l * HID, h, tmpf, nullptr, N);
    k_reduce_sums<<<1024, 256, 0, stream>>>(tmpf, red, NB);
    k_final_stats<<<1, 256, 0, stream>>>(red, stats, 1024, 1.f / (float)NB);
    k_ln_silu<<<(NB + 255) / 256, 256, 0, stream>>>(tmpf, stats, ln_w + l * HID,
                                                    ln_b + l * HID, h, hbf, NB);
  }

  // attention pool (query collapses to one vector since q0 is the broadcast mean)
  k_hmean<<<HID, 256, 0, stream>>>(h, hmean, N);
  k_qvec<<<1, 128, 0, stream>>>(hmean, wq, bq, qvec);
  k_kq<<<1, 512, 0, stream>>>(wk, bk, qvec, kqv, kqb);
  k_qk<<<(N * 4 + 255) / 256, 256, 0, stream>>>(h, kqv, kqb, qk, N);
  k_gemm128<false, false, true, false, true, false><<<tiles, 256, 0, stream>>>(
      hbf, wvbf, bv, nullptr, Vbuf, nullptr, N);
  k_attn_pool<<<256, 128, 0, stream>>>(sub_masks, qk, Vbuf, sub, N);
  k_out_proj<<<256, 128, 0, stream>>>(sub, wo, bo, sgq, sub2, sg);
  k_topk_out<<<1, 128, 0, stream>>>(sub2, sg, (float*)d_out);
}
